// K2Layer_25494925869275
// MI455X (gfx1250) — compile-verified, matrix operand order checked by reading, not run on hardware
//
#include <hip/hip_runtime.h>
#include <hip/hip_bf16.h>
#include <math.h>

typedef __bf16 bf16_t;
typedef __attribute__((ext_vector_type(16))) __bf16 v16bf;
typedef __attribute__((ext_vector_type(8)))  float  v8f;

static constexpr int Bb   = 8;
static constexpr int Ss   = 1024;
static constexpr int Dd   = 1024;
static constexpr int Rr   = 16;
static constexpr int NC   = 8;     // chunks per sequence
static constexpr int CL   = 128;   // chunk length
static constexpr int DFF  = 4096;
static constexpr int MTOT = Bb * Ss;

#define DEV __device__ __forceinline__

DEV float sigmoidf_(float x) { return 1.0f / (1.0f + expf(-x)); }

DEV unsigned short f2bfu(float f) {
    union { float f; unsigned u; } a; a.f = f;
    unsigned r = a.u + 0x7FFFu + ((a.u >> 16) & 1u);   // round-to-nearest-even
    return (unsigned short)(r >> 16);
}
DEV bf16_t f2bf(float f) {
    union { unsigned short s; bf16_t b; } o; o.s = f2bfu(f); return o.b;
}

// CDNA5 async global->LDS copy (tracked on ASYNCcnt, no VGPR round-trip).
// GVS mode: 64-bit uniform SGPR base + per-lane signed 32-bit byte offset.
// First operand = per-lane LDS byte address (low 32 bits of generic shared ptr).
DEV void g2l_async_b128(unsigned lds_off, const void* sbase, int goff) {
    asm volatile("global_load_async_to_lds_b128 %0, %1, %2"
                 :: "v"(lds_off), "v"(goff), "s"(sbase)
                 : "memory");
}
DEV void wait_async0() {
    asm volatile("s_wait_asynccnt 0" ::: "memory");
}

DEV float block_sum(float v, float* smem, int tid) {
    #pragma unroll
    for (int off = 16; off > 0; off >>= 1) v += __shfl_xor(v, off, 32);
    int wid = tid >> 5;
    if ((tid & 31) == 0) smem[wid] = v;
    __syncthreads();
    if (wid == 0) {
        float x = (tid < 8) ? smem[tid] : 0.0f;
        #pragma unroll
        for (int off = 4; off > 0; off >>= 1) x += __shfl_xor(x, off, 32);
        if (tid == 0) smem[0] = x;
    }
    __syncthreads();
    return smem[0];
}

// ---------------------------------------------------------------------------
// Kernel 1: rmsnorm1 + qk = h_norm @ [u,v] + l2norm(q)*alpha, l2norm(k)
// one block (256 threads) per (b,s) row
// ---------------------------------------------------------------------------
__global__ __launch_bounds__(256) void k_norm1_qk(
    const float* __restrict__ h, const float* __restrict__ u,
    const float* __restrict__ v, const float* __restrict__ alpha_logit,
    const float* __restrict__ norm1_w,
    float* __restrict__ hn, float* __restrict__ qa, float* __restrict__ kk)
{
    __shared__ float sred[32];
    __shared__ float swc[8 * 33];
    __shared__ float sqk[36];
    int row = blockIdx.x, tid = threadIdx.x;

    float4 hv = ((const float4*)(h + (size_t)row * Dd))[tid];
    float ss  = hv.x*hv.x + hv.y*hv.y + hv.z*hv.z + hv.w*hv.w;
    float tot = block_sum(ss, sred, tid);
    float sc  = rsqrtf(tot * (1.0f / Dd) + 1e-6f);
    float4 wv = ((const float4*)norm1_w)[tid];
    float hns[4] = { hv.x*sc*wv.x, hv.y*sc*wv.y, hv.z*sc*wv.z, hv.w*sc*wv.w };
    float4 hno = { hns[0], hns[1], hns[2], hns[3] };
    ((float4*)(hn + (size_t)row * Dd))[tid] = hno;

    float acc[32];
    #pragma unroll
    for (int c = 0; c < 32; c++) acc[c] = 0.0f;
    int dbase = tid * 4;
    #pragma unroll
    for (int j = 0; j < 4; j++) {
        int d = dbase + j;
        float hvj = hns[j];
        const float4* up = (const float4*)(u + (size_t)d * Rr);
        const float4* vp = (const float4*)(v + (size_t)d * Rr);
        #pragma unroll
        for (int q4 = 0; q4 < 4; q4++) {
            float4 uu = up[q4];
            acc[q4*4+0] += hvj*uu.x; acc[q4*4+1] += hvj*uu.y;
            acc[q4*4+2] += hvj*uu.z; acc[q4*4+3] += hvj*uu.w;
            float4 vv = vp[q4];
            acc[16+q4*4+0] += hvj*vv.x; acc[16+q4*4+1] += hvj*vv.y;
            acc[16+q4*4+2] += hvj*vv.z; acc[16+q4*4+3] += hvj*vv.w;
        }
    }
    int lane = tid & 31, wid = tid >> 5;
    #pragma unroll
    for (int c = 0; c < 32; c++) {
        float x = acc[c];
        #pragma unroll
        for (int off = 16; off > 0; off >>= 1) x += __shfl_xor(x, off, 32);
        if (lane == 0) swc[wid * 33 + c] = x;
    }
    __syncthreads();
    if (tid < 32) {
        float x = 0.0f;
        #pragma unroll
        for (int w = 0; w < 8; w++) x += swc[w * 33 + tid];
        sqk[tid] = x;
    }
    __syncthreads();
    if (tid == 0) {
        float qn = 0.0f, kn = 0.0f;
        #pragma unroll
        for (int r = 0; r < 16; r++) { qn += sqk[r]*sqk[r]; kn += sqk[16+r]*sqk[16+r]; }
        sqk[32] = fmaxf(sqrtf(qn), 1e-8f);
        sqk[33] = fmaxf(sqrtf(kn), 1e-8f);
    }
    __syncthreads();
    if (tid < 16) {
        float alpha = sigmoidf_(alpha_logit[tid]);          // ALPHA_CAP = 1
        qa[(size_t)row * Rr + tid] = sqk[tid] / sqk[32] * alpha;
    } else if (tid < 32) {
        kk[(size_t)row * Rr + (tid - 16)] = sqk[tid] / sqk[33];
    }
}

// ---------------------------------------------------------------------------
// Decay scan, pass 1: per-chunk local final states (zero init)
// grid (Dd/256, NC, Bb)
// ---------------------------------------------------------------------------
__global__ __launch_bounds__(256) void k_scan_local(
    const float* __restrict__ hn, const float* __restrict__ kk,
    const float* __restrict__ decay_logit, float* __restrict__ SL)
{
    __shared__ float kbuf[32 * 16];
    int tid = threadIdx.x;
    int d = blockIdx.x * 256 + tid, c = blockIdx.y, b = blockIdx.z;
    float gamma[Rr], st[Rr];
    #pragma unroll
    for (int r = 0; r < Rr; r++) {
        gamma[r] = fminf(fmaxf(sigmoidf_(decay_logit[r]), 0.15f), 1.0f);
        st[r] = 0.0f;
    }
    int s0 = c * CL;
    for (int t = 0; t < CL / 32; t++) {
        __syncthreads();
        if (tid < 128)
            ((float4*)kbuf)[tid] =
                ((const float4*)(kk + (size_t)(b * Ss + s0 + t * 32) * Rr))[tid];
        __syncthreads();
        #pragma unroll 4
        for (int i = 0; i < 32; i++) {
            float hval = hn[((size_t)b * Ss + (s0 + t * 32 + i)) * Dd + d];
            #pragma unroll
            for (int r = 0; r < Rr; r++)
                st[r] = gamma[r] * st[r] + kbuf[i * 16 + r] * hval;
        }
    }
    float* out = SL + ((size_t)(b * NC + c) * Rr) * Dd + d;
    #pragma unroll
    for (int r = 0; r < Rr; r++) out[(size_t)r * Dd] = st[r];
}

// ---------------------------------------------------------------------------
// Decay scan, pass 2: combine chunk carries; rewrite SL with chunk INIT states
// ---------------------------------------------------------------------------
__global__ __launch_bounds__(256) void k_scan_carry(
    const float* __restrict__ decay_logit, float* __restrict__ SL)
{
    int idx = blockIdx.x * 256 + threadIdx.x;  // over Bb*Rr*Dd
    int d = idx % Dd, r = (idx / Dd) % Rr, b = idx / (Dd * Rr);
    float gamma = fminf(fmaxf(sigmoidf_(decay_logit[r]), 0.15f), 1.0f);
    float g128 = powf(gamma, (float)CL);
    float C = 0.0f;
    for (int c = 0; c < NC; c++) {
        float* p = SL + ((size_t)(b * NC + c) * Rr + r) * Dd + d;
        float tmp = *p;
        *p = C;                 // initial state entering chunk c
        C = g128 * C + tmp;
    }
}

// ---------------------------------------------------------------------------
// Decay scan, pass 3: full scan with init state + fused causal conv
// writes mix = gate*conv(h_norm) + decay_out   (bf16, A-operand of proj GEMM)
// ---------------------------------------------------------------------------
__global__ __launch_bounds__(256) void k_scan_out(
    const float* __restrict__ hn, const float* __restrict__ kk,
    const float* __restrict__ qa, const float* __restrict__ decay_logit,
    const float* __restrict__ gate_logit, const float* __restrict__ conv_k,
    const float* __restrict__ SL, bf16_t* __restrict__ mixbf)
{
    __shared__ float kbuf[32 * 16];
    __shared__ float qbuf[32 * 16];
    int tid = threadIdx.x;
    int d = blockIdx.x * 256 + tid, c = blockIdx.y, b = blockIdx.z;
    float gamma[Rr], st[Rr];
    const float* ip = SL + ((size_t)(b * NC + c) * Rr) * Dd + d;
    #pragma unroll
    for (int r = 0; r < Rr; r++) {
        gamma[r] = fminf(fmaxf(sigmoidf_(decay_logit[r]), 0.15f), 1.0f);
        st[r] = ip[(size_t)r * Dd];
    }
    float gate = sigmoidf_(gate_logit[0]);
    float c0 = conv_k[0], c1 = conv_k[1], c2 = conv_k[2], c3 = conv_k[3];
    int s0 = c * CL;
    const float* hrow = hn + (size_t)b * Ss * Dd + d;
    float hm1 = (s0 >= 1) ? hrow[(size_t)(s0 - 1) * Dd] : 0.0f;
    float hm2 = (s0 >= 2) ? hrow[(size_t)(s0 - 2) * Dd] : 0.0f;
    float hm3 = (s0 >= 3) ? hrow[(size_t)(s0 - 3) * Dd] : 0.0f;

    for (int t = 0; t < CL / 32; t++) {
        __syncthreads();
        size_t base = (size_t)(b * Ss + s0 + t * 32) * Rr;
        if (tid < 128)      ((float4*)kbuf)[tid]       = ((const float4*)(kk + base))[tid];
        else                ((float4*)qbuf)[tid - 128] = ((const float4*)(qa + base))[tid - 128];
        __syncthreads();
        #pragma unroll 4
        for (int i = 0; i < 32; i++) {
            int s = s0 + t * 32 + i;
            float hval = hrow[(size_t)s * Dd];
            float out = 0.0f;
            #pragma unroll
            for (int r = 0; r < Rr; r++) {
                st[r] = gamma[r] * st[r] + kbuf[i * 16 + r] * hval;
                out += qbuf[i * 16 + r] * st[r];
            }
            float conv = c0 * hval + c1 * hm1 + c2 * hm2 + c3 * hm3;
            hm3 = hm2; hm2 = hm1; hm1 = hval;
            mixbf[((size_t)b * Ss + s) * Dd + d] = f2bf(gate * conv + out);
        }
    }
}

// ---------------------------------------------------------------------------
// rmsnorm2: h2 -> hn2 (bf16)
// ---------------------------------------------------------------------------
__global__ __launch_bounds__(256) void k_norm2(
    const float* __restrict__ h2, const float* __restrict__ norm2_w,
    bf16_t* __restrict__ hn2)
{
    __shared__ float sred[32];
    int row = blockIdx.x, tid = threadIdx.x;
    float4 hv = ((const float4*)(h2 + (size_t)row * Dd))[tid];
    float tot = block_sum(hv.x*hv.x + hv.y*hv.y + hv.z*hv.z + hv.w*hv.w, sred, tid);
    float sc = rsqrtf(tot * (1.0f / Dd) + 1e-6f);
    float4 wv = ((const float4*)norm2_w)[tid];
    unsigned lo = (unsigned)f2bfu(hv.x*sc*wv.x) | ((unsigned)f2bfu(hv.y*sc*wv.y) << 16);
    unsigned hi = (unsigned)f2bfu(hv.z*sc*wv.z) | ((unsigned)f2bfu(hv.w*sc*wv.w) << 16);
    ((uint2*)(hn2 + (size_t)row * Dd))[tid] = make_uint2(lo, hi);
}

// ---------------------------------------------------------------------------
// fp32 -> bf16 conversion (weights)
// ---------------------------------------------------------------------------
__global__ __launch_bounds__(256) void k_cvt_bf16(
    const float* __restrict__ in, bf16_t* __restrict__ out, int n)
{
    int i = blockIdx.x * 256 + threadIdx.x;
    if (i < n) out[i] = f2bf(in[i]);
}

// ---------------------------------------------------------------------------
// WMMA bf16 GEMM: C = A[MxK] * B + bias (+residual | gelu)
//   B_IS_NK=1: B stored [N][K] (Y = X @ W^T, proj case)
//   B_IS_NK=0: B stored [K][N] (Y = X @ W,   mlp case)
//   EPI=0: outF = resid + (acc + bias)       (fp32)
//   EPI=1: outB = gelu_tanh(acc + bias)      (bf16)
// 128x128 tile, BK=32, 256 threads = 8 waves (4m x 2n), 2x4 WMMA accs/wave.
// Double-buffered LDS; contiguous tiles prefetched with
// global_load_async_to_lds_b128 (ASYNCcnt) one K-step ahead of the WMMAs.
// ---------------------------------------------------------------------------
template <int B_IS_NK, int EPI, int Nc, int Kc>
__global__ __launch_bounds__(256) void k_gemm(
    const bf16_t* __restrict__ A, const bf16_t* __restrict__ Bm,
    const float* __restrict__ bias, const float* __restrict__ resid,
    float* __restrict__ outF, bf16_t* __restrict__ outB)
{
    constexpr int LDT  = 40;                      // 80B row stride (16B-aligned)
    constexpr int TILE = 128 * LDT;
    __shared__ bf16_t As[2 * TILE];
    __shared__ bf16_t Bs[2 * TILE];
    int tid = threadIdx.x;
    int n0 = blockIdx.x * 128, m0 = blockIdx.y * 128;
    int lane = tid & 31, wid = tid >> 5;
    int wm = wid >> 1, wn = wid & 1;
    int koff = (lane >= 16) ? 8 : 0;              // ISA 16-bit A/B fragment layout

    // async staging of the contiguous operands (A always; B in NK layout)
    auto stage_async = [&](int kt, int buf) {
        bf16_t* Ad = As + buf * TILE;
        const void* abase = (const void*)(A + (size_t)m0 * Kc + kt);
        #pragma unroll
        for (int l = 0; l < 2; l++) {
            int idx = tid + l * 256;
            int rowi = idx >> 2, cq = idx & 3;
            g2l_async_b128((unsigned)(size_t)&Ad[rowi * LDT + cq * 8],
                           abase, rowi * (Kc * 2) + cq * 16);
        }
        if (B_IS_NK) {
            bf16_t* Bd = Bs + buf * TILE;
            const void* bbase = (const void*)(Bm + (size_t)n0 * Kc + kt);
            #pragma unroll
            for (int l = 0; l < 2; l++) {
                int idx = tid + l * 256;
                int rowi = idx >> 2, cq = idx & 3;
                g2l_async_b128((unsigned)(size_t)&Bd[rowi * LDT + cq * 8],
                               bbase, rowi * (Kc * 2) + cq * 16);
            }
        }
    };
    // B in [K][N] layout: global loads now, LDS transpose-scatter later
    auto stage_load_b = [&](int kt, uint4* bv) {
        #pragma unroll
        for (int l = 0; l < 2; l++) {
            int idx = tid + l * 256;
            int kr = idx >> 4, nq = idx & 15;
            bv[l] = *(const uint4*)(Bm + (size_t)(kt + kr) * Nc + n0 + nq * 8);
        }
    };
    auto stage_store_b = [&](int buf, const uint4* bv) {
        bf16_t* Bd = Bs + buf * TILE;
        #pragma unroll
        for (int l = 0; l < 2; l++) {
            int idx = tid + l * 256;
            int kr = idx >> 4, nq = idx & 15;
            const bf16_t* vp = (const bf16_t*)&bv[l];
            #pragma unroll
            for (int j = 0; j < 8; j++)
                Bd[(nq * 8 + j) * LDT + kr] = vp[j];
        }
    };

    v8f acc[2][4];
    #pragma unroll
    for (int sm = 0; sm < 2; sm++)
        #pragma unroll
        for (int sn = 0; sn < 4; sn++) {
            v8f z = {0.f,0.f,0.f,0.f,0.f,0.f,0.f,0.f};
            acc[sm][sn] = z;
        }

    // pipeline prologue: tile 0 -> buffer 0
    stage_async(0, 0);
    if (!B_IS_NK) {
        uint4 bv[2];
        stage_load_b(0, bv);
        stage_store_b(0, bv);
    }
    wait_async0();
    __syncthreads();

    for (int kt = 0, t = 0; kt < Kc; kt += 32, t ^= 1) {
        bool pf = (kt + 32 < Kc);
        uint4 bv[2];
        if (pf) {
            stage_async(kt + 32, t ^ 1);          // async prefetch next tile
            if (!B_IS_NK) stage_load_b(kt + 32, bv);
        }

        const bf16_t* Ac = As + t * TILE;
        const bf16_t* Bc = Bs + t * TILE;
        union Frag { v16bf v; uint4 q[2]; };
        Frag af[2], bfg[4];
        #pragma unroll
        for (int sm = 0; sm < 2; sm++) {
            int ar = wm * 32 + sm * 16 + (lane & 15);
            af[sm].q[0] = *(const uint4*)(&Ac[ar * LDT + koff]);
            af[sm].q[1] = *(const uint4*)(&Ac[ar * LDT + koff + 16]);
        }
        #pragma unroll
        for (int sn = 0; sn < 4; sn++) {
            int br = wn * 64 + sn * 16 + (lane & 15);
            bfg[sn].q[0] = *(const uint4*)(&Bc[br * LDT + koff]);
            bfg[sn].q[1] = *(const uint4*)(&Bc[br * LDT + koff + 16]);
        }
        #pragma unroll
        for (int sm = 0; sm < 2; sm++)
            #pragma unroll
            for (int sn = 0; sn < 4; sn++)
                acc[sm][sn] = __builtin_amdgcn_wmma_f32_16x16x32_bf16(
                    false, af[sm].v, false, bfg[sn].v,
                    (short)0, acc[sm][sn], false, false);

        if (pf && !B_IS_NK) stage_store_b(t ^ 1, bv);  // behind the WMMAs
        wait_async0();
        __syncthreads();
    }

    // epilogue: lane L holds (M = g + 8*(L/16), N = L%16) of each 16x16 tile
    #pragma unroll
    for (int sm = 0; sm < 2; sm++) {
        #pragma unroll
        for (int sn = 0; sn < 4; sn++) {
            int colg = n0 + wn * 64 + sn * 16 + (lane & 15);
            int row0 = m0 + wm * 32 + sm * 16 + 8 * (lane >> 4);
            float bsv = bias[colg];
            if (EPI == 0) {
                const float* rp = resid + (size_t)row0 * Nc + colg;
                float* op = outF + (size_t)row0 * Nc + colg;
                float rv[8];
                #pragma unroll
                for (int g = 0; g < 8; g++) rv[g] = rp[g * Nc];
                #pragma unroll
                for (int g = 0; g < 8; g++)
                    op[g * Nc] = rv[g] + acc[sm][sn][g] + bsv;
            } else {
                bf16_t* op = outB + (size_t)row0 * Nc + colg;
                #pragma unroll
                for (int g = 0; g < 8; g++) {
                    float x = acc[sm][sn][g] + bsv;
                    float tt = tanhf(0.7978845608028654f * (x + 0.044715f * x * x * x));
                    op[g * Nc] = f2bf(0.5f * x * (1.0f + tt));
                }
            }
        }
    }
}

// ---------------------------------------------------------------------------
extern "C" void kernel_launch(void* const* d_in, const int* in_sizes, int n_in,
                              void* d_out, int out_size, void* d_ws, size_t ws_size,
                              hipStream_t stream)
{
    const float* h           = (const float*)d_in[0];
    const float* u           = (const float*)d_in[1];
    const float* v           = (const float*)d_in[2];
    const float* decay_logit = (const float*)d_in[3];
    const float* alpha_logit = (const float*)d_in[4];
    const float* gate_logit  = (const float*)d_in[5];
    const float* conv_k      = (const float*)d_in[6];
    const float* proj_w      = (const float*)d_in[7];
    const float* proj_b      = (const float*)d_in[8];
    const float* norm1_w     = (const float*)d_in[9];
    const float* norm2_w     = (const float*)d_in[10];
    const float* mlp_w1      = (const float*)d_in[11];
    const float* mlp_b1      = (const float*)d_in[12];
    const float* mlp_w2      = (const float*)d_in[13];
    const float* mlp_b2      = (const float*)d_in[14];
    (void)in_sizes; (void)n_in; (void)out_size; (void)ws_size;

    char* ws = (char*)d_ws;
    size_t off = 0;
    auto alloc = [&](size_t bytes) -> char* {
        char* p = ws + off;
        off += (bytes + 255) & ~(size_t)255;
        return p;
    };
    float*  hn    = (float*)alloc((size_t)MTOT * Dd * 4);
    float*  qa    = (float*)alloc((size_t)MTOT * Rr * 4);
    float*  kk    = (float*)alloc((size_t)MTOT * Rr * 4);
    float*  SL    = (float*)alloc((size_t)Bb * NC * Rr * Dd * 4);
    bf16_t* mixbf = (bf16_t*)alloc((size_t)MTOT * Dd * 2);
    float*  h2    = (float*)alloc((size_t)MTOT * Dd * 4);
    bf16_t* hn2   = (bf16_t*)alloc((size_t)MTOT * Dd * 2);
    bf16_t* G     = (bf16_t*)alloc((size_t)MTOT * DFF * 2);
    bf16_t* pwb   = (bf16_t*)alloc((size_t)Dd * Dd * 2);
    bf16_t* w1b   = (bf16_t*)alloc((size_t)Dd * DFF * 2);
    bf16_t* w2b   = (bf16_t*)alloc((size_t)DFF * Dd * 2);

    // weight conversions fp32 -> bf16
    k_cvt_bf16<<<(Dd * Dd + 255) / 256, 256, 0, stream>>>(proj_w, pwb, Dd * Dd);
    k_cvt_bf16<<<(Dd * DFF + 255) / 256, 256, 0, stream>>>(mlp_w1, w1b, Dd * DFF);
    k_cvt_bf16<<<(DFF * Dd + 255) / 256, 256, 0, stream>>>(mlp_w2, w2b, DFF * Dd);

    // rmsnorm1 + q/k projection + l2norm
    k_norm1_qk<<<MTOT, 256, 0, stream>>>(h, u, v, alpha_logit, norm1_w, hn, qa, kk);

    // chunk-parallel decay scan + fused causal conv -> mix (bf16)
    dim3 gscan(Dd / 256, NC, Bb);
    k_scan_local<<<gscan, 256, 0, stream>>>(hn, kk, decay_logit, SL);
    k_scan_carry<<<(Bb * Rr * Dd) / 256, 256, 0, stream>>>(decay_logit, SL);
    k_scan_out<<<gscan, 256, 0, stream>>>(hn, kk, qa, decay_logit, gate_logit,
                                          conv_k, SL, mixbf);

    // proj: h2 = h + mix @ proj_w^T + proj_b
    dim3 gp(Dd / 128, MTOT / 128);
    k_gemm<1, 0, Dd, Dd><<<gp, 256, 0, stream>>>(mixbf, pwb, proj_b, h, h2, nullptr);

    // rmsnorm2
    k_norm2<<<MTOT, 256, 0, stream>>>(h2, norm2_w, hn2);

    // mlp1: G = gelu(hn2 @ w1 + b1)   (bf16)
    dim3 g1(DFF / 128, MTOT / 128);
    k_gemm<0, 1, DFF, Dd><<<g1, 256, 0, stream>>>(hn2, w1b, mlp_b1, nullptr,
                                                  nullptr, G);

    // mlp2: out = h2 + G @ w2 + b2    (fp32 -> d_out)
    dim3 g2(Dd / 128, MTOT / 128);
    k_gemm<0, 0, Dd, DFF><<<g2, 256, 0, stream>>>(G, w2b, mlp_b2, h2,
                                                  (float*)d_out, nullptr);
}
